// LSTM_25555055412080
// MI455X (gfx1250) — matrix operand order repaired and checked by reading, not compiled, not run on hardware
//
#include <hip/hip_runtime.h>

// ---------------------------------------------------------------------------
// Fused 2-layer LSTM (B=256, T=1024, H=256) for MI455X (gfx1250, wave32).
// Persistent cooperative kernel, bf16 WMMA 16x16x32 with f32 accumulate,
// weights resident in LDS, double-buffered h-states in L2-resident workspace,
// grid barrier (atomics + s_sleep) per timestep, layers software-pipelined.
// Round-2 tweaks: K loop unrolled 2x (8 static WMMAs), bias/input-proj
// weights hoisted to registers outside the 1024-step serial loop.
// ---------------------------------------------------------------------------

typedef __attribute__((ext_vector_type(16))) __bf16 v16bf;
typedef __attribute__((ext_vector_type(8)))  float  v8f;

#define B_SZ 256
#define T_SZ 1024
#define H_SZ 256
#define NWG_TOT 64        // 32 WGs for layer0 + 32 WGs for layer1

// workspace layout (bytes)
#define OFF_H1   256                              // 2 x [B,H] bf16 double buffer (layer0 h / layer1 x)
#define OFF_H2   (OFF_H1 + 2*B_SZ*H_SZ*2)         // 2 x [B,H] bf16 double buffer (layer1 h)
#define OFF_C0   (OFF_H2 + 2*B_SZ*H_SZ*2)         // [B,H] f32 cell state layer0
#define OFF_C2   (OFF_C0 + B_SZ*H_SZ*4)           // [B,H] f32 cell state layer1
#define WS_TOTAL (OFF_C2 + B_SZ*H_SZ*4)           // ~1.0 MB

__device__ __forceinline__ float fsig(float x)  { return 1.0f / (1.0f + __expf(-x)); }
__device__ __forceinline__ float ftanh(float x) { return 2.0f * fsig(2.0f * x) - 1.0f; }

// A fragment (16x32 bf16): per lane two 16B chunks at +0 and +16 elements.
__device__ __forceinline__ v16bf loadA(const __bf16* p) {
    union { v16bf v; uint4 q[2]; } u;
    u.q[0] = *reinterpret_cast<const uint4*>(p);
    u.q[1] = *reinterpret_cast<const uint4*>(p + 16);
    return u.v;
}
// B fragment (32x16 bf16): per lane two 16B chunks at +0 and +16 elements from LDS.
__device__ __forceinline__ v16bf loadB(const __bf16* p) {
    union { v16bf v; uint4 q[2]; } u;
    u.q[0] = *reinterpret_cast<const uint4*>(p);
    u.q[1] = *reinterpret_cast<const uint4*>(p + 16);
    return u.v;
}

__device__ __forceinline__ void gridbar(unsigned* cnt, volatile unsigned* gen, unsigned nwg) {
    __threadfence();
    __syncthreads();
    if (threadIdx.x == 0) {
        unsigned g = *gen;
        if (atomicAdd(cnt, 1u) == nwg - 1u) {
            *cnt = 0u;
            __threadfence();
            *gen = g + 1u;
        } else {
            while (*gen == g) __builtin_amdgcn_s_sleep(2);
        }
    }
    __syncthreads();
    __threadfence();
}

__global__ void init_ws_kernel(uint4* p, int n) {
    uint4 z = make_uint4(0u, 0u, 0u, 0u);
    for (int i = blockIdx.x * blockDim.x + threadIdx.x; i < n; i += gridDim.x * blockDim.x)
        p[i] = z;
}

// ---------------------------------------------------------------------------
// Persistent fused LSTM kernel. 64 WGs x 256 threads (8 wave32s).
// Layer L WG grid: 4 (B blocks of 64) x 8 (h-col blocks of 32).
// Wave w: row-block rb=w>>1 (16 B rows), h-set hs=w&1 (16 h cols),
//         computes i/f/g/o 16x16 accumulators for its (rb,hs) tile.
// LDS W layout: row r = q*32 + j  ->  global gate row q*H + j0 + j, K-major.
// Layer1 K = 512: k<256 from own h2 state, k>=256 from layer0 output h1.
// ---------------------------------------------------------------------------
__global__ __launch_bounds__(256)
void lstm_fused_kernel(const float* __restrict__ xi, const float* __restrict__ yi,
                       const float* __restrict__ zi,
                       const float* __restrict__ w_ih0, const float* __restrict__ w_hh0,
                       const float* __restrict__ b_ih0, const float* __restrict__ b_hh0,
                       const float* __restrict__ w_ih1, const float* __restrict__ w_hh1,
                       const float* __restrict__ b_ih1, const float* __restrict__ b_hh1,
                       __bf16* __restrict__ h1b, __bf16* __restrict__ h2b,
                       float* __restrict__ c0buf, float* __restrict__ c2buf,
                       unsigned* __restrict__ ctrl) {
    __shared__ __align__(16) __bf16 sW[128 * 512];   // 128 KB (layer1), layer0 uses 64 KB
    __shared__ float sBias[128];
    __shared__ float sXw[128 * 3];                   // layer0 input-proj weights
    __shared__ float sX[3][64];                      // staged x/y/z for this WG's B rows

    const int tid   = threadIdx.x;
    const int layer = blockIdx.x >> 5;               // 0 or 1
    const int wg    = blockIdx.x & 31;
    const int B0    = (wg & 3) * 64;
    const int j0    = (wg >> 2) * 32;
    const int KW    = layer ? 512 : 256;

    // ---- stage weights + bias into LDS (once) ----
    for (int idx = tid; idx < 128 * KW; idx += 256) {
        int r = idx / KW, k = idx - r * KW;
        int grow = (r >> 5) * H_SZ + j0 + (r & 31);
        float wv;
        if (layer == 0)      wv = w_hh0[grow * H_SZ + k];
        else if (k < H_SZ)   wv = w_hh1[grow * H_SZ + k];
        else                 wv = w_ih1[grow * H_SZ + (k - H_SZ)];
        sW[r * KW + k] = (__bf16)wv;
    }
    if (tid < 128) {
        int grow = (tid >> 5) * H_SZ + j0 + (tid & 31);
        sBias[tid] = layer ? (b_ih1[grow] + b_hh1[grow]) : (b_ih0[grow] + b_hh0[grow]);
        if (layer == 0) {
            sXw[tid * 3 + 0] = w_ih0[grow * 3 + 0];
            sXw[tid * 3 + 1] = w_ih0[grow * 3 + 1];
            sXw[tid * 3 + 2] = w_ih0[grow * 3 + 2];
        }
    }
    __syncthreads();

    const int lane   = tid & 31;
    const int w      = tid >> 5;
    const int rb     = w >> 1;
    const int hs     = w & 1;
    const int n      = lane & 15;
    const int laneHi = lane >> 4;                    // 0 or 1
    const int arow   = B0 + rb * 16 + n;             // A-matrix row (= batch index)
    const int hcol   = j0 + hs * 16 + n;             // output h column

    // ---- hoist step-invariant per-lane constants out of the serial loop ----
    float bi[4], xw[4][3];
#pragma unroll
    for (int q = 0; q < 4; ++q) {
        int r = q * 32 + hs * 16 + n;
        bi[q] = sBias[r];
        if (layer == 0) {
            xw[q][0] = sXw[r * 3 + 0];
            xw[q][1] = sXw[r * 3 + 1];
            xw[q][2] = sXw[r * 3 + 2];
        } else {
            xw[q][0] = xw[q][1] = xw[q][2] = 0.0f;
        }
    }

    float* cbuf = layer ? c2buf : c0buf;
    unsigned* cnt = ctrl;
    volatile unsigned* gen = ctrl + 1;
    const unsigned nwg = gridDim.x;

    for (int s = 0; s <= T_SZ; ++s) {
        bool active;
        const __bf16* hread = nullptr;   // K 0..255 source (own recurrent h)
        const __bf16* xread = nullptr;   // K 256..511 source (layer1 input = h1)
        __bf16* hwrite = nullptr;

        if (layer == 0) {
            active = (s < T_SZ);
            if (active) {
                if (tid < 64) {
                    int b = B0 + tid;
                    sX[0][tid] = xi[b * T_SZ + s];
                    sX[1][tid] = yi[b * T_SZ + s];
                    sX[2][tid] = zi[b * T_SZ + s];
                }
                __syncthreads();
                hread  = h1b + ((s + 1) & 1) * (B_SZ * H_SZ);
                hwrite = h1b + (s & 1) * (B_SZ * H_SZ);
            }
        } else {
            active = (s >= 1);
            if (active) {
                int tl = s - 1;
                hread  = h2b + ((tl + 1) & 1) * (B_SZ * H_SZ);
                xread  = h1b + (tl & 1) * (B_SZ * H_SZ);
                hwrite = h2b + (tl & 1) * (B_SZ * H_SZ);
            }
        }

        if (active) {
            // ---- GEMM: gates[16 B rows x (4 gates x 16 cols)] via bf16 WMMA ----
            // K unrolled 2x: two independent 32-deep WMMA blocks per iteration.
            v8f acc[4] = {v8f{}, v8f{}, v8f{}, v8f{}};
            for (int kk = 0; kk < KW; kk += 64) {
#pragma unroll
                for (int half = 0; half < 2; ++half) {
                    const int k0 = kk + half * 32;
                    const __bf16* asrc = (k0 < H_SZ) ? hread : xread;
                    const int kl = (k0 < H_SZ) ? k0 : (k0 - H_SZ);
                    v16bf a = loadA(asrc + (size_t)arow * H_SZ + kl + laneHi * 8);
#pragma unroll
                    for (int q = 0; q < 4; ++q) {
                        const __bf16* bp =
                            &sW[(q * 32 + hs * 16 + n) * KW + k0 + laneHi * 8];
                        v16bf bfrag = loadB(bp);
                        acc[q] = __builtin_amdgcn_wmma_f32_16x16x32_bf16(
                            false, a, false, bfrag, (short)0, acc[q], false, false);
                    }
                }
            }

            // ---- gate nonlinearity + cell/h update (wave-local: owns i,f,g,o) ----
#pragma unroll
            for (int m = 0; m < 8; ++m) {
                int bl = rb * 16 + m + laneHi * 8;       // local B row (D-layout M)
                float g4[4];
#pragma unroll
                for (int q = 0; q < 4; ++q) g4[q] = acc[q][m] + bi[q];
                if (layer == 0) {
                    float xv = sX[0][bl], yv = sX[1][bl], zv = sX[2][bl];
#pragma unroll
                    for (int q = 0; q < 4; ++q)
                        g4[q] += xw[q][0] * xv + xw[q][1] * yv + xw[q][2] * zv;
                }
                size_t idx = (size_t)(B0 + bl) * H_SZ + hcol;
                float cold = cbuf[idx];
                float iv = fsig(g4[0]), fv = fsig(g4[1]);
                float gv = ftanh(g4[2]), ov = fsig(g4[3]);
                float cn = fv * cold + iv * gv;
                cbuf[idx] = cn;
                hwrite[idx] = (__bf16)(ov * ftanh(cn));
            }
        }

        gridbar(cnt, gen, nwg);
    }
}

// Final dense head: out[b,:] = h2_last[b,:] @ dense_w^T + dense_b  (tiny)
__global__ void dense_kernel(const __bf16* __restrict__ h2, const float* __restrict__ wd,
                             const float* __restrict__ bd, float* __restrict__ out) {
    int b = threadIdx.x;
    float a0 = bd[0], a1 = bd[1], a2 = bd[2];
    for (int k = 0; k < H_SZ; ++k) {
        float hv = (float)h2[b * H_SZ + k];
        a0 += hv * wd[0 * H_SZ + k];
        a1 += hv * wd[1 * H_SZ + k];
        a2 += hv * wd[2 * H_SZ + k];
    }
    out[b * 3 + 0] = a0;
    out[b * 3 + 1] = a1;
    out[b * 3 + 2] = a2;
}

extern "C" void kernel_launch(void* const* d_in, const int* in_sizes, int n_in,
                              void* d_out, int out_size, void* d_ws, size_t ws_size,
                              hipStream_t stream) {
    const float* xi    = (const float*)d_in[0];
    const float* yi    = (const float*)d_in[1];
    const float* zi    = (const float*)d_in[2];
    const float* w_ih0 = (const float*)d_in[3];
    const float* w_hh0 = (const float*)d_in[4];
    const float* b_ih0 = (const float*)d_in[5];
    const float* b_hh0 = (const float*)d_in[6];
    const float* w_ih1 = (const float*)d_in[7];
    const float* w_hh1 = (const float*)d_in[8];
    const float* b_ih1 = (const float*)d_in[9];
    const float* b_hh1 = (const float*)d_in[10];
    const float* dw    = (const float*)d_in[11];
    const float* db    = (const float*)d_in[12];

    char* ws = (char*)d_ws;
    unsigned* ctrl = (unsigned*)ws;
    __bf16* h1b = (__bf16*)(ws + OFF_H1);
    __bf16* h2b = (__bf16*)(ws + OFF_H2);
    float*  c0  = (float*)(ws + OFF_C0);
    float*  c2  = (float*)(ws + OFF_C2);

    // zero control block + states + double buffers (deterministic per call)
    init_ws_kernel<<<64, 256, 0, stream>>>((uint4*)d_ws, WS_TOTAL / 16);

    lstm_fused_kernel<<<NWG_TOT, 256, 0, stream>>>(
        xi, yi, zi, w_ih0, w_hh0, b_ih0, b_hh0,
        w_ih1, w_hh1, b_ih1, b_hh1, h1b, h2b, c0, c2, ctrl);

    // t = T-1 = 1023 was written to h2 parity buffer 1
    dense_kernel<<<1, B_SZ, 0, stream>>>(h2b + (size_t)(B_SZ * H_SZ), dw, db, (float*)d_out);
}
